// BayesianLayer_38405597561399
// MI455X (gfx1250) — matrix operand order repaired and checked by reading, not compiled, not run on hardware
//
#include <hip/hip_runtime.h>

// ---------------------------------------------------------------------------
// Bayesian linear layer for MI455X (gfx1250, wave32, WMMA):
//   W  = w_mu + softplus(w_rho) * w_eps      (prep kernel, stored bf16)
//   b  = b_mu + softplus(b_rho) * b_eps      (prep kernel, f32)
//   Y  = dropout( X @ W^T + b )              (bf16 WMMA GEMM, fused epilogue)
//
// GEMM: 128x128 block tile, 8 wave32s (4M x 2N), 32x64 per wave (2x4 frags),
// K staged through double-buffered LDS in steps of 32 (one WMMA K).
//  - X tile: global_load_b128 -> v_cvt_pk_bf16_f32 -> ds_store_b128 (needs cvt)
//  - W tile: GLOBAL_LOAD_ASYNC_TO_LDS_B128 (ASYNCcnt path, bypasses VGPRs)
// One barrier per K-stage; staging overlaps the 8 WMMAs.
// ---------------------------------------------------------------------------

typedef __attribute__((ext_vector_type(16))) __bf16 v16bf;
typedef __attribute__((ext_vector_type(8)))  __bf16 v8bf;
typedef __attribute__((ext_vector_type(8)))  float  v8f;
typedef __attribute__((ext_vector_type(4)))  float  v4f;

#define B_ROWS     65536
#define K_DIM      512
#define N_DIM      512
#define DROP_P     0.2f
#define DROP_SCALE 1.25f   // 1 / (1 - DROP_P)

#define BM  128            // block tile M (rows of X)
#define BN  128            // block tile N (output features)
#define BK  32             // K step == WMMA bf16 K
#define LDA 40             // padded LDS row stride (80B; 20-bank stride, conflict-free)
#define NSTAGE (K_DIM / BK)

#define MFRAG 2            // 16x16 fragments per wave, M dir (wave tile 32 rows)
#define NFRAG 4            // 16x16 fragments per wave, N dir (wave tile 64 cols)

// ---------------------------------------------------------------------------
// Kernel 1: reparameterize weights -> bf16, bias -> f32 (tiny: 262k elems)
// ---------------------------------------------------------------------------
__global__ __launch_bounds__(256) void bayes_prep(
    const float* __restrict__ w_mu, const float* __restrict__ w_rho,
    const float* __restrict__ w_eps, const float* __restrict__ b_mu,
    const float* __restrict__ b_rho, const float* __restrict__ b_eps,
    __bf16* __restrict__ Wbf, float* __restrict__ bias)
{
    const int tid = blockIdx.x * 256 + threadIdx.x;
    if (tid < N_DIM * K_DIM) {
        const float rho = w_rho[tid];
        const float sp  = (rho > 15.0f) ? rho : log1pf(__expf(rho)); // softplus
        const float w   = w_mu[tid] + sp * w_eps[tid];
        Wbf[tid] = (__bf16)w;
    }
    if (tid < N_DIM) {
        const float rho = b_rho[tid];
        const float sp  = (rho > 15.0f) ? rho : log1pf(__expf(rho));
        bias[tid] = b_mu[tid] + sp * b_eps[tid];
    }
}

// Async global->LDS copy of 2x16B per lane (tracked by ASYNCcnt).
// INST_OFFSET is added to both the LDS and global addresses, so the two
// chunks share operands and differ only in offset.
__device__ __forceinline__ void async_copy_32B(unsigned lds_addr, const void* gptr)
{
    asm volatile("global_load_async_to_lds_b128 %0, %1, off"
                 :: "v"(lds_addr), "v"(gptr) : "memory");
    asm volatile("global_load_async_to_lds_b128 %0, %1, off offset:16"
                 :: "v"(lds_addr), "v"(gptr) : "memory");
}
__device__ __forceinline__ void wait_async0()
{
    asm volatile("s_wait_asynccnt 0x0" ::: "memory");
}

// ---------------------------------------------------------------------------
// Kernel 2: Y = dropout(X @ W^T + b). Grid: (65536/128) x (512/128) = 512 x 4.
// ---------------------------------------------------------------------------
__global__ __launch_bounds__(256) void bayes_gemm(
    const float*  __restrict__ X,      // [B_ROWS][K_DIM] f32
    const __bf16* __restrict__ Wbf,    // [N_DIM][K_DIM]  bf16 (row n contiguous in K)
    const float*  __restrict__ bias,   // [N_DIM] f32
    const float*  __restrict__ dropU,  // [B_ROWS][N_DIM] f32 uniforms
    float*        __restrict__ Y)      // [B_ROWS][N_DIM] f32
{
    __shared__ __bf16 As[2][BM][LDA];  // X tiles (double-buffered)
    __shared__ __bf16 Bs[2][BN][LDA];  // W tiles (double-buffered)

    const int t     = threadIdx.x;
    const int lane  = t & 31;          // wave32 lane
    const int wave  = t >> 5;          // 0..7
    const int waveM = wave >> 1;       // 0..3 -> 32-row strip
    const int waveN = wave & 1;        // 0..1 -> 64-col strip

    const int rowBlk = blockIdx.x * BM;
    const int colBlk = blockIdx.y * BN;

    // Cooperative tile-load assignment: 256 threads cover
    //   X tile 128x32 f32  (16 f32 / thread  = 4 x b128)
    //   W tile 128x32 bf16 (32 B / thread    = 2 x async b128)
    const int xr  = t & 127;           // tile row
    const int xcg = t >> 7;            // 0..1 : 16-float K chunk
    const int wr  = t & 127;
    const int wcg = t >> 7;            // 0..1 : 16-bf16 K chunk

    const float*  xBase = X   + (size_t)(rowBlk + xr) * K_DIM + xcg * 16;
    const __bf16* wBase = Wbf + (size_t)(colBlk + wr) * K_DIM + wcg * 16;

    // Per-lane LDS byte addresses for the async W-tile copies (both buffers).
    const unsigned bsDst0 = (unsigned)(unsigned long long)&Bs[0][wr][wcg * 16];
    const unsigned bsDst1 = (unsigned)(unsigned long long)&Bs[1][wr][wcg * 16];

    // Fragment lane decomposition (ISA 16-bit A / B layouts, wave32)
    const int fr  = lane & 15;         // A: row M within fragment;  B: column N
    const int fhi = lane >> 4;         // half-wave selector

    v8f acc[MFRAG][NFRAG] = {};        // 2x4 fragments of 16x16 f32

    // ---- prologue: stage 0 into LDS buffer 0 ----
    {
        async_copy_32B(bsDst0, wBase);                    // W tile, ASYNCcnt
        const v4f* xp = (const v4f*)xBase;
        const v4f f0 = xp[0], f1 = xp[1], f2 = xp[2], f3 = xp[3];
        v8bf p0, p1;
        #pragma unroll
        for (int i = 0; i < 4; ++i) {
            p0[i] = (__bf16)f0[i]; p0[4 + i] = (__bf16)f1[i];
            p1[i] = (__bf16)f2[i]; p1[4 + i] = (__bf16)f3[i];
        }
        *(v8bf*)&As[0][xr][xcg * 16]     = p0;
        *(v8bf*)&As[0][xr][xcg * 16 + 8] = p1;
        wait_async0();
    }
    __syncthreads();

    int cur = 0;
    for (int s = 0; s < NSTAGE; ++s) {
        const bool hasNext = (s + 1 < NSTAGE);

        // ---- issue next stage's loads first (overlap with WMMAs) ----
        v4f f0, f1, f2, f3;
        if (hasNext) {
            async_copy_32B(cur ? bsDst0 : bsDst1,          // W tile -> other buffer
                           wBase + (s + 1) * BK);
            const v4f* xp = (const v4f*)(xBase + (s + 1) * BK);
            f0 = xp[0]; f1 = xp[1]; f2 = xp[2]; f3 = xp[3];
            if (s + 2 < NSTAGE)
                __builtin_prefetch(xBase + (s + 2) * BK, 0, 0); // global_prefetch_b8
        }

        // ---- gather fragments from LDS buffer `cur` ----
        // A (16x32 bf16): lanes 0-15 hold K 0-7 / 16-23 of row M=fr;
        //                 lanes 16-31 hold K 8-15 / 24-31.
        v16bf a[MFRAG], b[NFRAG];
        #pragma unroll
        for (int mf = 0; mf < MFRAG; ++mf) {
            const __bf16* ap = &As[cur][waveM * 32 + mf * 16 + fr][fhi * 8];
            const v8bf lo = *(const v8bf*)ap;                 // ds_load_b128
            const v8bf hi = *(const v8bf*)(ap + 16);
            a[mf] = __builtin_shufflevector(lo, hi, 0,1,2,3,4,5,6,7,8,9,10,11,12,13,14,15);
        }
        // B (32x16 bf16): lane holds column N=fr; lanes 0-15 K 0-7 / 16-23, lanes 16-31 K 8-15 / 24-31.
        #pragma unroll
        for (int nf = 0; nf < NFRAG; ++nf) {
            const __bf16* bp = &Bs[cur][waveN * 64 + nf * 16 + fr][fhi * 8];
            const v8bf lo = *(const v8bf*)bp;
            const v8bf hi = *(const v8bf*)(bp + 16);
            b[nf] = __builtin_shufflevector(lo, hi, 0,1,2,3,4,5,6,7,8,9,10,11,12,13,14,15);
        }

        // ---- 8 WMMAs per wave per K-stage ----
        #pragma unroll
        for (int mf = 0; mf < MFRAG; ++mf)
            #pragma unroll
            for (int nf = 0; nf < NFRAG; ++nf)
                acc[mf][nf] = __builtin_amdgcn_wmma_f32_16x16x32_bf16(
                    /*neg_a=*/false, a[mf], /*neg_b=*/false, b[nf],
                    /*c_mod=*/(short)0, acc[mf][nf],
                    /*reuse_a=*/false, /*reuse_b=*/false);

        // ---- convert + store next X tile into the other LDS buffer ----
        if (hasNext) {
            const int nxt = cur ^ 1;
            v8bf p0, p1;
            #pragma unroll
            for (int i = 0; i < 4; ++i) {
                p0[i] = (__bf16)f0[i]; p0[4 + i] = (__bf16)f1[i];
                p1[i] = (__bf16)f2[i]; p1[4 + i] = (__bf16)f3[i];
            }
            *(v8bf*)&As[nxt][xr][xcg * 16]     = p0;
            *(v8bf*)&As[nxt][xr][xcg * 16 + 8] = p1;
            wait_async0();                 // W tile for `nxt` landed in LDS
        }
        __syncthreads();   // single barrier per K-stage (ping-pong buffers)
        cur ^= 1;
    }

    // ---- fused epilogue: bias add + inverted dropout, straight to HBM ----
    // C/D layout: VGPR r -> M = r (lanes 0-15) / r+8 (lanes 16-31), N = lane&15.
    #pragma unroll
    for (int mf = 0; mf < MFRAG; ++mf) {
        #pragma unroll
        for (int nf = 0; nf < NFRAG; ++nf) {
            const int   col  = colBlk + waveN * 64 + nf * 16 + fr;
            const float bv   = bias[col];
            const int   row0 = rowBlk + waveM * 32 + mf * 16 + fhi * 8;
            #pragma unroll
            for (int r = 0; r < 8; ++r) {
                const size_t idx = (size_t)(row0 + r) * N_DIM + col;
                const float  u   = dropU[idx];
                const float  y   = acc[mf][nf][r] + bv;
                Y[idx] = (u >= DROP_P) ? y * DROP_SCALE : 0.0f;
            }
        }
    }
}

// ---------------------------------------------------------------------------
// Host launcher. Input order: x, w_mu, w_rho, b_mu, b_rho, w_eps, b_eps, drop_u
// Workspace: [ Wbf16: 512*512*2 B ][ bias: 512*4 B ]  (~514 KB)
// ---------------------------------------------------------------------------
extern "C" void kernel_launch(void* const* d_in, const int* in_sizes, int n_in,
                              void* d_out, int out_size, void* d_ws, size_t ws_size,
                              hipStream_t stream)
{
    const float* x     = (const float*)d_in[0];
    const float* w_mu  = (const float*)d_in[1];
    const float* w_rho = (const float*)d_in[2];
    const float* b_mu  = (const float*)d_in[3];
    const float* b_rho = (const float*)d_in[4];
    const float* w_eps = (const float*)d_in[5];
    const float* b_eps = (const float*)d_in[6];
    const float* dropu = (const float*)d_in[7];
    float* out = (float*)d_out;

    __bf16* Wbf  = (__bf16*)d_ws;
    float*  bias = (float*)((char*)d_ws + (size_t)N_DIM * K_DIM * sizeof(__bf16));

    bayes_prep<<<dim3((N_DIM * K_DIM + 255) / 256), dim3(256), 0, stream>>>(
        w_mu, w_rho, w_eps, b_mu, b_rho, b_eps, Wbf, bias);

    bayes_gemm<<<dim3(B_ROWS / BM, N_DIM / BN), dim3(256), 0, stream>>>(
        x, Wbf, bias, dropu, out);
}
